// Head_48232482734463
// MI455X (gfx1250) — matrix operand order repaired and checked by reading, not compiled, not run on hardware
//
#include <hip/hip_runtime.h>
#include <hip/hip_bf16.h>

typedef __attribute__((ext_vector_type(16))) _Float16 v16h;
typedef __attribute__((ext_vector_type(8)))  _Float16 v8h;
typedef __attribute__((ext_vector_type(8)))  float    v8f;

#define B_DIM 8
#define T_DIM 2048
#define C_DIM 1024
#define H_DIM 128

static __device__ __forceinline__ v8f wmma_f16(const v16h& a, const v16h& b, const v8f& c) {
    return __builtin_amdgcn_wmma_f32_16x16x32_f16(false, a, false, b, (short)0, c, false, false);
}

// ---------------------------------------------------------------------------
// Kernel 0: transpose + downconvert weights: Wt[mat][h][c] = (f16) W_mat[c*H + h]
// ---------------------------------------------------------------------------
__global__ void head_wt_kernel(const float* __restrict__ Wq,
                               const float* __restrict__ Wk,
                               const float* __restrict__ Wv,
                               _Float16* __restrict__ Wt) {
    int tid = blockIdx.x * blockDim.x + threadIdx.x;           // 0 .. 3*H*C-1
    int mat = tid / (H_DIM * C_DIM);
    int rem = tid % (H_DIM * C_DIM);
    int h = rem / C_DIM;
    int c = rem % C_DIM;
    const float* W = (mat == 0) ? Wq : (mat == 1) ? Wk : Wv;
    Wt[tid] = (_Float16)W[c * H_DIM + h];
}

// ---------------------------------------------------------------------------
// Kernel 1: QKV projection. One wave (32 threads) per (16-row tile, matrix).
//   Q,K stored f16 [B*T][H]; V stored transposed f16 [B][H][T].
// ---------------------------------------------------------------------------
__global__ void head_qkv_kernel(const float* __restrict__ x,
                                const _Float16* __restrict__ Wt,
                                const float* __restrict__ bq,
                                const float* __restrict__ bk,
                                const float* __restrict__ bv,
                                _Float16* __restrict__ Qf,
                                _Float16* __restrict__ Kf,
                                _Float16* __restrict__ Vt) {
    const int lane = threadIdx.x;         // 0..31
    const int m    = lane & 15;
    const int half = lane >> 4;
    const int tile = blockIdx.x;          // 0 .. (B*T/16 - 1)
    const int mat  = blockIdx.y;          // 0=Q 1=K 2=V

    const float* xrow = x + (size_t)(tile * 16 + m) * C_DIM;   // A row for this lane
    const _Float16* Wb = Wt + (size_t)mat * H_DIM * C_DIM;

    v8f acc[8] = {};
    for (int kc = 0; kc < C_DIM / 32; ++kc) {
        // A fragment (16x32 f16): lane holds row M=m; K = half*16 + e
        v16h a;
        #pragma unroll
        for (int e = 0; e < 8; ++e) {
            a[e]     = (_Float16)xrow[kc * 32 + half * 16 + e];
            a[8 + e] = (_Float16)xrow[kc * 32 + half * 16 + 8 + e];
        }
        #pragma unroll
        for (int nt = 0; nt < 8; ++nt) {
            // B fragment (32x16): lane col n = nt*16+m; K = half*16 + e (contiguous in Wt)
            const _Float16* wp = Wb + (size_t)(nt * 16 + m) * C_DIM + kc * 32 + half * 16;
            v16h bf = *(const v16h*)wp;
            acc[nt] = wmma_f16(a, bf, acc[nt]);
        }
    }

    const float* bias = (mat == 0) ? bq : (mat == 1) ? bk : bv;
    #pragma unroll
    for (int nt = 0; nt < 8; ++nt) {
        int n = nt * 16 + m;
        float bb = bias[n];
        #pragma unroll
        for (int r = 0; r < 8; ++r) {
            int grow = tile * 16 + r + 8 * half;               // global row in [B*T]
            float val = acc[nt][r] + bb;
            if (mat == 0) {
                Qf[(size_t)grow * H_DIM + n] = (_Float16)val;
            } else if (mat == 1) {
                Kf[(size_t)grow * H_DIM + n] = (_Float16)val;
            } else {
                int bidx = grow >> 11;                          // T = 2048
                int t    = grow & (T_DIM - 1);
                Vt[((size_t)bidx * H_DIM + n) * T_DIM + t] = (_Float16)val;
            }
        }
    }
}

// ---------------------------------------------------------------------------
// Kernel 2: flash attention. One wave per (batch, 16-query tile).
// ---------------------------------------------------------------------------
__global__ void head_attn_kernel(const _Float16* __restrict__ Qf,
                                 const _Float16* __restrict__ Kf,
                                 const _Float16* __restrict__ Vt,
                                 float* __restrict__ out) {
    const int lane = threadIdx.x;
    const int m    = lane & 15;
    const int half = lane >> 4;
    const int b    = blockIdx.y;
    const int q0   = blockIdx.x * 16;

    const _Float16* Qb = Qf + (size_t)b * T_DIM * H_DIM;
    const _Float16* Kb = Kf + (size_t)b * T_DIM * H_DIM;
    const _Float16* Vb = Vt + (size_t)b * H_DIM * T_DIM;

    __shared__ _Float16 pls[16][32];       // P tile staging (C-layout -> A-layout)

    // Q fragments: 4 chunks of K=32 over H=128
    v16h qa[4];
    const _Float16* qp = Qb + (size_t)(q0 + m) * H_DIM;
    #pragma unroll
    for (int c4 = 0; c4 < 4; ++c4) {
        v8h lo = *(const v8h*)(qp + c4 * 32 + half * 16);
        v8h hi = *(const v8h*)(qp + c4 * 32 + half * 16 + 8);
        #pragma unroll
        for (int e = 0; e < 8; ++e) { qa[c4][e] = lo[e]; qa[c4][8 + e] = hi[e]; }
    }

    v8f o[8] = {};
    float mrow[8], lrow[8];
    #pragma unroll
    for (int r = 0; r < 8; ++r) { mrow[r] = -1e30f; lrow[r] = 0.0f; }

    const float scale = 0.03125f;          // C_DIM^-0.5 = 1/32 (reference uses embed dim)
    const int nblocks = (q0 + 16 + 31) / 32;

    for (int sb = 0; sb < nblocks; ++sb) {
        const int s0 = sb * 32;

        // ---- S = Q @ K^T for 32 keys (two 16-key tiles) ----
        v8f sA = {}, sB = {};
        #pragma unroll
        for (int c4 = 0; c4 < 4; ++c4) {
            v16h kb1 = *(const v16h*)(Kb + (size_t)(s0 +      m) * H_DIM + c4 * 32 + half * 16);
            v16h kb2 = *(const v16h*)(Kb + (size_t)(s0 + 16 + m) * H_DIM + c4 * 32 + half * 16);
            sA = wmma_f16(qa[c4], kb1, sA);
            sB = wmma_f16(qa[c4], kb2, sB);
        }

        // ---- online softmax (row = r + 8*half, col = m in C-layout) ----
        #pragma unroll
        for (int r = 0; r < 8; ++r) {
            int qidx = q0 + r + 8 * half;
            float v1 = sA[r] * scale; if (s0 +      m > qidx) v1 = -1e30f;
            float v2 = sB[r] * scale; if (s0 + 16 + m > qidx) v2 = -1e30f;

            float mx = fmaxf(v1, v2);
            #pragma unroll
            for (int off = 1; off < 16; off <<= 1)
                mx = fmaxf(mx, __shfl_xor(mx, off, 32));

            float mnew = fmaxf(mrow[r], mx);
            float corr = __expf(mrow[r] - mnew);
            float p1 = __expf(v1 - mnew);
            float p2 = __expf(v2 - mnew);
            float ps = p1 + p2;
            #pragma unroll
            for (int off = 1; off < 16; off <<= 1)
                ps += __shfl_xor(ps, off, 32);

            lrow[r] = lrow[r] * corr + ps;
            mrow[r] = mnew;
            #pragma unroll
            for (int nt = 0; nt < 8; ++nt) o[nt][r] *= corr;

            pls[r + 8 * half][m]      = (_Float16)p1;
            pls[r + 8 * half][16 + m] = (_Float16)p2;
        }
        // single wave: same-wave LDS ops are in-order (ISA 7.3); compiler adds waits

        // ---- P (16x32) as A-fragment ----
        v16h pa;
        #pragma unroll
        for (int e = 0; e < 8; ++e) {
            pa[e]     = pls[m][half * 16 + e];
            pa[8 + e] = pls[m][half * 16 + 8 + e];
        }

        // ---- O += P @ V  (Vt is [H][T] so B-fragments are contiguous) ----
        #pragma unroll
        for (int nt = 0; nt < 8; ++nt) {
            v16h vb = *(const v16h*)(Vb + (size_t)(nt * 16 + m) * T_DIM + s0 + half * 16);
            o[nt] = wmma_f16(pa, vb, o[nt]);
        }
    }

    // ---- write out[b][q][h] = o / l ----
    #pragma unroll
    for (int r = 0; r < 8; ++r) {
        float inv = 1.0f / lrow[r];
        size_t row = (size_t)b * T_DIM + q0 + r + 8 * half;
        #pragma unroll
        for (int nt = 0; nt < 8; ++nt) {
            out[row * H_DIM + nt * 16 + m] = o[nt][r] * inv;
        }
    }
}

// ---------------------------------------------------------------------------
extern "C" void kernel_launch(void* const* d_in, const int* in_sizes, int n_in,
                              void* d_out, int out_size, void* d_ws, size_t ws_size,
                              hipStream_t stream) {
    const float* x  = (const float*)d_in[0];
    const float* Wq = (const float*)d_in[1];
    const float* bq = (const float*)d_in[2];
    const float* Wk = (const float*)d_in[3];
    const float* bk = (const float*)d_in[4];
    const float* Wv = (const float*)d_in[5];
    const float* bv = (const float*)d_in[6];
    float* out = (float*)d_out;

    // workspace layout (f16): Qf 4MB | Kf 4MB | Vt 4MB | Wt 0.75MB
    char* ws = (char*)d_ws;
    const size_t QKV_BYTES = (size_t)B_DIM * T_DIM * H_DIM * sizeof(_Float16);
    _Float16* Qf = (_Float16*)(ws);
    _Float16* Kf = (_Float16*)(ws + QKV_BYTES);
    _Float16* Vt = (_Float16*)(ws + 2 * QKV_BYTES);
    _Float16* Wt = (_Float16*)(ws + 3 * QKV_BYTES);

    // 0) weight transpose + f16 downconvert
    {
        int total = 3 * H_DIM * C_DIM;
        head_wt_kernel<<<total / 256, 256, 0, stream>>>(Wq, Wk, Wv, Wt);
    }
    // 1) QKV projection (one wave per 16-row tile per matrix)
    {
        dim3 grid((B_DIM * T_DIM) / 16, 3);
        head_qkv_kernel<<<grid, 32, 0, stream>>>(x, Wt, bq, bk, bv, Qf, Kf, Vt);
    }
    // 2) causal flash attention (one wave per (batch, 16-query tile))
    {
        dim3 grid(T_DIM / 16, B_DIM);
        head_attn_kernel<<<grid, 32, 0, stream>>>(Qf, Kf, Vt, out);
    }
}
